// RNN_GRU2_1589137900342
// MI455X (gfx1250) — matrix operand order repaired and checked by reading, not compiled, not run on hardware
//
#include <hip/hip_runtime.h>
#include <hip/hip_bf16.h>
#include <math.h>

#define B_ 64
#define T_ 512
#define D_ 512
#define H_ 1024

typedef __bf16 bf16_t;
typedef __attribute__((ext_vector_type(16))) __bf16 v16bf;
typedef __attribute__((ext_vector_type(8)))  __bf16 v8bf;
typedef __attribute__((ext_vector_type(8)))  float  v8f;

// ---------------------------------------------------------------------------
// WMMA fragment loaders (bf16 16x16x32, wave32).
// A (16x32 MxK): lanes 0-15 -> M=lane, elems 0..7 = K 0..7, 8..15 = K 16..23
//                lanes16-31 -> M=lane-16, elems 0..7 = K 8..15, 8..15 = K 24..31
// B (32x16 KxN): lanes 0-15 -> N=lane, elems 0..7 = K 0..7, 8..15 = K 16..23
//                lanes16-31 -> N=lane-16, elems 0..7 = K 8..15, 8..15 = K 24..31
// ---------------------------------------------------------------------------
static __device__ __forceinline__ v16bf load_a_frag(const bf16_t* base, int ld,
                                                    int m0, int kb, int lane) {
  const int m    = lane & 15;
  const int koff = (lane & 16) ? 8 : 0;
  const bf16_t* p = base + (size_t)(m0 + m) * ld + kb + koff;
  v8bf lo = *(const v8bf*)(p);
  v8bf hi = *(const v8bf*)(p + 16);
  v16bf r;
#pragma unroll
  for (int i = 0; i < 8; ++i) { r[i] = lo[i]; r[i + 8] = hi[i]; }
  return r;
}

static __device__ __forceinline__ v16bf load_b_frag(const bf16_t* base, int ld,
                                                    int n0, int kb, int lane) {
  const int n    = lane & 15;
  const int koff = (lane & 16) ? 8 : 0;
  const bf16_t* p = base + (size_t)(n0 + n) * ld + kb + koff;
  v8bf lo = *(const v8bf*)(p);
  v8bf hi = *(const v8bf*)(p + 16);
  v16bf r;
#pragma unroll
  for (int i = 0; i < 8; ++i) { r[i] = lo[i]; r[i + 8] = hi[i]; }
  return r;
}

static __device__ __forceinline__ v8f wmma_bf16(v16bf a, v16bf b, v8f c) {
  return __builtin_amdgcn_wmma_f32_16x16x32_bf16(false, a, false, b,
                                                 (short)0, c, false, false);
}

static __device__ __forceinline__ float sigmoidf_(float x) {
  return 1.0f / (1.0f + __expf(-x));
}

// ---------------------------------------------------------------------------
// fp32 -> bf16 conversion pre-pass
// ---------------------------------------------------------------------------
__global__ void cvt_f32_bf16_kernel(const float* __restrict__ src,
                                    bf16_t* __restrict__ dst, size_t n) {
  size_t i = (size_t)blockIdx.x * blockDim.x + threadIdx.x;
  size_t stride = (size_t)gridDim.x * blockDim.x;
  for (; i < n; i += stride) dst[i] = (bf16_t)src[i];
}

__global__ void zero_h_kernel(float* __restrict__ hstate,
                              bf16_t* __restrict__ hbf) {
  int i = blockIdx.x * blockDim.x + threadIdx.x;  // exactly B_*H_ threads
  hstate[i] = 0.0f;
  hbf[i]    = (bf16_t)0.0f;
}

// ---------------------------------------------------------------------------
// Input projections (register-blocked 32x64 per wave):
//   xproj[t,b,0:3H] = x[b,t,:] @ W3^T + bias3
// A = xbf [B*T, D] (m = b*T + t), B = W3 [3H, D].
// Tiles: M 2048/2=1024, N 192/4=48 -> 49152 waves -> 6144 blocks of 8 waves.
// ---------------------------------------------------------------------------
__global__ void xproj_kernel(const bf16_t* __restrict__ xbf,
                             const bf16_t* __restrict__ W3,
                             const float* __restrict__ buz,
                             const float* __restrict__ bur,
                             const float* __restrict__ bu,
                             float* __restrict__ xproj) {
  const int lane = threadIdx.x & 31;
  const int wv   = blockIdx.x * 8 + (threadIdx.x >> 5);
  const int NTB  = 48;                 // N super-tiles of 64
  const int m0   = (wv / NTB) * 32;
  const int n0   = (wv % NTB) * 64;

  v8f c00 = {}, c01 = {}, c02 = {}, c03 = {};
  v8f c10 = {}, c11 = {}, c12 = {}, c13 = {};
  for (int k = 0; k < D_; k += 32) {
    v16bf a0 = load_a_frag(xbf, D_, m0,      k, lane);
    v16bf a1 = load_a_frag(xbf, D_, m0 + 16, k, lane);
    v16bf b0 = load_b_frag(W3, D_, n0,      k, lane);
    v16bf b1 = load_b_frag(W3, D_, n0 + 16, k, lane);
    v16bf b2 = load_b_frag(W3, D_, n0 + 32, k, lane);
    v16bf b3 = load_b_frag(W3, D_, n0 + 48, k, lane);
    c00 = wmma_bf16(a0, b0, c00);  c01 = wmma_bf16(a0, b1, c01);
    c02 = wmma_bf16(a0, b2, c02);  c03 = wmma_bf16(a0, b3, c03);
    c10 = wmma_bf16(a1, b0, c10);  c11 = wmma_bf16(a1, b1, c11);
    c12 = wmma_bf16(a1, b2, c12);  c13 = wmma_bf16(a1, b3, c13);
  }

  const int moff = (lane & 16) ? 8 : 0;
  const int ncol = lane & 15;
  const v8f* accs[2][4] = {{&c00, &c01, &c02, &c03}, {&c10, &c11, &c12, &c13}};
#pragma unroll
  for (int mt = 0; mt < 2; ++mt) {
#pragma unroll
    for (int nt = 0; nt < 4; ++nt) {
      const int j = n0 + nt * 16 + ncol;
      const float bias =
          (j < H_) ? buz[j] : (j < 2 * H_) ? bur[j - H_] : bu[j - 2 * H_];
      const v8f& acc = *accs[mt][nt];
#pragma unroll
      for (int v = 0; v < 8; ++v) {
        int m = m0 + mt * 16 + moff + v;
        int b = m >> 9;          // / T_
        int t = m & (T_ - 1);
        xproj[((size_t)t * B_ + b) * (3 * H_) + j] = acc[v] + bias;
      }
    }
  }
}

// ---------------------------------------------------------------------------
// Recurrent phase 1: zr_pre = xproj(z,r) + h @ Wzr^T + b ; z=sigmoid, rh=r*h
// M=64, N=2048, K=1024. 1 tile/wave for max parallelism (512 waves).
// ---------------------------------------------------------------------------
__global__ void gru_phase1_kernel(const bf16_t* __restrict__ hbf,
                                  const bf16_t* __restrict__ Wzr,
                                  const float* __restrict__ bwz,
                                  const float* __restrict__ bwr,
                                  const float* __restrict__ xproj,
                                  const float* __restrict__ hstate,
                                  float* __restrict__ zbuf,
                                  bf16_t* __restrict__ rhbf,
                                  int t) {
  const int lane = threadIdx.x & 31;
  const int wv   = blockIdx.x * 8 + (threadIdx.x >> 5);
  const int NT   = (2 * H_) / 16;  // 128
  const int m0   = (wv / NT) * 16;
  const int n0   = (wv % NT) * 16;

  v8f acc = {};
  for (int k = 0; k < H_; k += 32) {
    v16bf a = load_a_frag(hbf, H_, m0, k, lane);
    v16bf b = load_b_frag(Wzr, H_, n0, k, lane);
    acc = wmma_bf16(a, b, acc);
  }

  const int j = n0 + (lane & 15);
  const float* xp = xproj + (size_t)t * B_ * (3 * H_);
  const int moff  = (lane & 16) ? 8 : 0;
  if (j < H_) {                       // z half (wave-uniform)
    const float bias = bwz[j];
#pragma unroll
    for (int v = 0; v < 8; ++v) {
      int b = m0 + moff + v;
      float pre = acc[v] + bias + xp[(size_t)b * (3 * H_) + j];
      zbuf[b * H_ + j] = sigmoidf_(pre);
    }
  } else {                            // r half
    const int jj = j - H_;
    const float bias = bwr[jj];
#pragma unroll
    for (int v = 0; v < 8; ++v) {
      int b = m0 + moff + v;
      float pre = acc[v] + bias + xp[(size_t)b * (3 * H_) + j];
      float r = sigmoidf_(pre);
      rhbf[b * H_ + jj] = (bf16_t)(r * hstate[b * H_ + jj]);
    }
  }
}

// ---------------------------------------------------------------------------
// Recurrent phase 2: c = tanh(xproj_c + rh @ Ww^T + bw); h = z*h + (1-z)*c
// M=64, N=1024, K=1024. 256 waves.
// ---------------------------------------------------------------------------
__global__ void gru_phase2_kernel(const bf16_t* __restrict__ rhbf,
                                  const bf16_t* __restrict__ Wwb,
                                  const float* __restrict__ bw,
                                  const float* __restrict__ xproj,
                                  const float* __restrict__ zbuf,
                                  float* __restrict__ hstate,
                                  bf16_t* __restrict__ hbf,
                                  bf16_t* __restrict__ hsbf,
                                  int t) {
  const int lane = threadIdx.x & 31;
  const int wv   = blockIdx.x * 8 + (threadIdx.x >> 5);
  const int NT   = H_ / 16;  // 64
  const int m0   = (wv / NT) * 16;
  const int n0   = (wv % NT) * 16;

  v8f acc = {};
  for (int k = 0; k < H_; k += 32) {
    v16bf a = load_a_frag(rhbf, H_, m0, k, lane);
    v16bf b = load_b_frag(Wwb, H_, n0, k, lane);
    acc = wmma_bf16(a, b, acc);
  }

  const int j = n0 + (lane & 15);
  const float bias = bw[j];
  const float* xp = xproj + (size_t)t * B_ * (3 * H_) + 2 * H_;
  const int moff  = (lane & 16) ? 8 : 0;
#pragma unroll
  for (int v = 0; v < 8; ++v) {
    int b = m0 + moff + v;
    float pre = acc[v] + bias + xp[(size_t)b * (3 * H_) + j];
    float c = tanhf(pre);
    float z = zbuf[b * H_ + j];
    float ho = hstate[b * H_ + j];
    float hn = z * ho + (1.0f - z) * c;
    hstate[b * H_ + j] = hn;
    bf16_t hb = (bf16_t)hn;
    hbf[b * H_ + j] = hb;
    hsbf[((size_t)t * B_ + b) * H_ + j] = hb;
  }
}

// ---------------------------------------------------------------------------
// Output projection (register-blocked 32x64 per wave):
//   out[b,t,d] = sigmoid(hs[t,b,:] @ Wo[d,:] + bo[d])
// M = T*B = 32768 (m = t*B + b), N = 512, K = 1024.
// Tiles: 1024 x 8 super-tiles -> 8192 waves -> 1024 blocks of 8 waves.
// ---------------------------------------------------------------------------
__global__ void outproj_kernel(const bf16_t* __restrict__ hsbf,
                               const bf16_t* __restrict__ Wob,
                               const float* __restrict__ bo,
                               float* __restrict__ out) {
  const int lane = threadIdx.x & 31;
  const int wv   = blockIdx.x * 8 + (threadIdx.x >> 5);
  const int NTB  = 8;                  // N super-tiles of 64
  const int m0   = (wv / NTB) * 32;
  const int n0   = (wv % NTB) * 64;

  v8f c00 = {}, c01 = {}, c02 = {}, c03 = {};
  v8f c10 = {}, c11 = {}, c12 = {}, c13 = {};
  for (int k = 0; k < H_; k += 32) {
    v16bf a0 = load_a_frag(hsbf, H_, m0,      k, lane);
    v16bf a1 = load_a_frag(hsbf, H_, m0 + 16, k, lane);
    v16bf b0 = load_b_frag(Wob, H_, n0,      k, lane);
    v16bf b1 = load_b_frag(Wob, H_, n0 + 16, k, lane);
    v16bf b2 = load_b_frag(Wob, H_, n0 + 32, k, lane);
    v16bf b3 = load_b_frag(Wob, H_, n0 + 48, k, lane);
    c00 = wmma_bf16(a0, b0, c00);  c01 = wmma_bf16(a0, b1, c01);
    c02 = wmma_bf16(a0, b2, c02);  c03 = wmma_bf16(a0, b3, c03);
    c10 = wmma_bf16(a1, b0, c10);  c11 = wmma_bf16(a1, b1, c11);
    c12 = wmma_bf16(a1, b2, c12);  c13 = wmma_bf16(a1, b3, c13);
  }

  const int moff = (lane & 16) ? 8 : 0;
  const int ncol = lane & 15;
  const v8f* accs[2][4] = {{&c00, &c01, &c02, &c03}, {&c10, &c11, &c12, &c13}};
#pragma unroll
  for (int mt = 0; mt < 2; ++mt) {
#pragma unroll
    for (int nt = 0; nt < 4; ++nt) {
      const int d = n0 + nt * 16 + ncol;
      const float bias = bo[d];
      const v8f& acc = *accs[mt][nt];
#pragma unroll
      for (int v = 0; v < 8; ++v) {
        int m = m0 + mt * 16 + moff + v;
        int t = m >> 6;          // / B_
        int b = m & (B_ - 1);
        out[((size_t)b * T_ + t) * D_ + d] = sigmoidf_(acc[v] + bias);
      }
    }
  }
}

// ---------------------------------------------------------------------------
// Host-side launcher
// ---------------------------------------------------------------------------
extern "C" void kernel_launch(void* const* d_in, const int* in_sizes, int n_in,
                              void* d_out, int out_size, void* d_ws, size_t ws_size,
                              hipStream_t stream) {
  const float* x   = (const float*)d_in[0];
  const float* Wuz = (const float*)d_in[1];
  const float* buz = (const float*)d_in[2];
  const float* Wwz = (const float*)d_in[3];
  const float* bwz = (const float*)d_in[4];
  const float* Wur = (const float*)d_in[5];
  const float* bur = (const float*)d_in[6];
  const float* Wwr = (const float*)d_in[7];
  const float* bwr = (const float*)d_in[8];
  const float* Wu  = (const float*)d_in[9];
  const float* bu  = (const float*)d_in[10];
  const float* Ww  = (const float*)d_in[11];
  const float* bw  = (const float*)d_in[12];
  const float* Wo  = (const float*)d_in[13];
  const float* bo  = (const float*)d_in[14];
  float* out = (float*)d_out;

  // Workspace carve-up (256B aligned)
  char* ws = (char*)d_ws;
  size_t off = 0;
  auto carve = [&](size_t bytes) -> void* {
    void* p = ws + off;
    off = (off + bytes + 255) & ~(size_t)255;
    return p;
  };
  bf16_t* xbf    = (bf16_t*)carve((size_t)B_ * T_ * D_ * 2);
  bf16_t* Wu3    = (bf16_t*)carve((size_t)3 * H_ * D_ * 2);
  bf16_t* Wzr    = (bf16_t*)carve((size_t)2 * H_ * H_ * 2);
  bf16_t* Wwb    = (bf16_t*)carve((size_t)H_ * H_ * 2);
  bf16_t* Wob    = (bf16_t*)carve((size_t)D_ * H_ * 2);
  float*  xproj  = (float*)carve((size_t)T_ * B_ * 3 * H_ * 4);
  bf16_t* hsbf   = (bf16_t*)carve((size_t)T_ * B_ * H_ * 2);
  float*  hstate = (float*)carve((size_t)B_ * H_ * 4);
  bf16_t* hbf    = (bf16_t*)carve((size_t)B_ * H_ * 2);
  float*  zbuf   = (float*)carve((size_t)B_ * H_ * 4);
  bf16_t* rhbf   = (bf16_t*)carve((size_t)B_ * H_ * 2);
  (void)ws_size; (void)in_sizes; (void)n_in; (void)out_size;

  auto cvt = [&](const float* s, bf16_t* d, size_t n) {
    int blocks = (int)((n + 1023) / 1024);
    if (blocks > 4096) blocks = 4096;
    cvt_f32_bf16_kernel<<<blocks, 256, 0, stream>>>(s, d, n);
  };

  // 1) bf16 conversions
  cvt(x,   xbf,                        (size_t)B_ * T_ * D_);
  cvt(Wuz, Wu3,                        (size_t)H_ * D_);
  cvt(Wur, Wu3 + (size_t)H_ * D_,      (size_t)H_ * D_);
  cvt(Wu,  Wu3 + (size_t)2 * H_ * D_,  (size_t)H_ * D_);
  cvt(Wwz, Wzr,                        (size_t)H_ * H_);
  cvt(Wwr, Wzr + (size_t)H_ * H_,      (size_t)H_ * H_);
  cvt(Ww,  Wwb,                        (size_t)H_ * H_);
  cvt(Wo,  Wob,                        (size_t)D_ * H_);

  // 2) zero hidden state (re-initialized every call — deterministic)
  zero_h_kernel<<<(B_ * H_) / 256, 256, 0, stream>>>(hstate, hbf);

  // 3) input projections: 1024 x 48 super-tiles = 49152 waves / 8 per block
  xproj_kernel<<<6144, 256, 0, stream>>>(xbf, Wu3, buz, bur, bu, xproj);

  // 4) recurrent scan: 512 steps, two kernels per step
  for (int t = 0; t < T_; ++t) {
    gru_phase1_kernel<<<64, 256, 0, stream>>>(hbf, Wzr, bwz, bwr, xproj,
                                              hstate, zbuf, rhbf, t);
    gru_phase2_kernel<<<32, 256, 0, stream>>>(rhbf, Wwb, bw, xproj, zbuf,
                                              hstate, hbf, hsbf, t);
  }

  // 5) output projection: 1024 x 8 super-tiles = 8192 waves / 8 per block
  outproj_kernel<<<1024, 256, 0, stream>>>(hsbf, Wob, bo, out);
}
